// TransformerBlock_29111288333146
// MI455X (gfx1250) — matrix operand order repaired and checked by reading, not compiled, not run on hardware
//
#include <hip/hip_runtime.h>
#include <math.h>
#include <stdint.h>

// Problem dims (fixed by reference)
static constexpr int Dc   = 1024;  // embed dim
static constexpr int BNc  = 4096;  // B*N rows
static constexpr int Hc   = 16;    // heads
static constexpr int HDc  = 64;    // head dim
static constexpr int Nseq = 2048;  // sequence length

typedef __attribute__((ext_vector_type(16))) _Float16 v16h;
typedef __attribute__((ext_vector_type(8)))  _Float16 v8h;
typedef __attribute__((ext_vector_type(8)))  float    v8f;
typedef __attribute__((ext_vector_type(4)))  unsigned int u32x4;
typedef __attribute__((ext_vector_type(8)))  int      i32x8;
typedef __attribute__((ext_vector_type(4)))  int      i32x4;

#if __has_builtin(__builtin_amdgcn_tensor_load_to_lds) && \
    __has_builtin(__builtin_amdgcn_s_wait_tensorcnt)
#define USE_TDM 1
#else
#define USE_TDM 0
#endif

// D = A(16x32) * B(32x16) + C, f16 inputs, f32 accum
__device__ __forceinline__ v8f wmma_f16(v16h a, v16h b, v8f c) {
  return __builtin_amdgcn_wmma_f32_16x16x32_f16(
      /*neg_a=*/false, a, /*neg_b=*/false, b,
      /*c_mod=*/(short)0, c, /*reuse_a=*/false, /*reuse_b=*/false);
}

// A-fragment loader: contiguous K order, a lane (row m = lane%16,
// base = 16*(lane/16)) holds K = base..base+15.
// `p` must already include the per-lane row offset and the 16*(lane/16) base.
__device__ __forceinline__ v16h ldA(const _Float16* p) {
  v8h lo = *(const v8h*)(p);
  v8h hi = *(const v8h*)(p + 8);
  return __builtin_shufflevector(lo, hi, 0,1,2,3,4,5,6,7,8,9,10,11,12,13,14,15);
}

#if USE_TDM
// Issue one TDM 2-D tile load: 64 rows x 128 bytes (row stride = Dc*2 bytes)
// from global `gsrc` into LDS byte address `lds_addr`. D# per CDNA5 ISA §8.
__device__ __forceinline__ void tdm_load_b_tile(const _Float16* gsrc,
                                                unsigned lds_addr) {
  const unsigned long long ga = (unsigned long long)(uintptr_t)gsrc;
  u32x4 g0;
  g0[0] = 1u;                                        // count=1, user D#
  g0[1] = lds_addr;                                  // LDS byte address
  g0[2] = (unsigned)(ga & 0xffffffffu);              // global_addr[31:0]
  g0[3] = (unsigned)((ga >> 32) & 0x01ffffffu)       // global_addr[56:32]
          | 0x80000000u;                             // type=2 ("image")
  i32x8 g1;
  g1[0] = (int)(3u << 16);          // wg_mask=0, data_size=3 (8B units)
  g1[1] = (int)(256u << 16);        // tensor_dim0 = 256 units (2KB row)
  g1[2] = (int)(4096u << 16);       // tensor_dim0 hi=0 | tensor_dim1 = 4096
  g1[3] = (int)(16u << 16);         // tensor_dim1 hi=0 | tile_dim0 = 16 (128B)
  g1[4] = 64;                       // tile_dim1 = 64 rows, tile_dim2 = 0
  g1[5] = 256;                      // tensor_dim0_stride = 256 units (2KB)
  g1[6] = 0;                        // stride0 hi | stride1 lo
  g1[7] = 0;                        // stride1 hi
  const i32x4 z4 = {0, 0, 0, 0};
#if defined(__clang_major__) && (__clang_major__ >= 23)
  const i32x8 z8 = {0, 0, 0, 0, 0, 0, 0, 0};
  __builtin_amdgcn_tensor_load_to_lds(g0, g1, z4, z4, z8, 0);
#else
  __builtin_amdgcn_tensor_load_to_lds(g0, g1, z4, z4, 0);
#endif
}
#endif

// ---------------------------------------------------------------------------
// Weight prep: f32 (d_in, d_out) -> f16 transposed (d_out, d_in)
// ---------------------------------------------------------------------------
__global__ __launch_bounds__(256) void w_transpose_f16(
    const float* __restrict__ W, _Float16* __restrict__ Wt) {
  __shared__ float tile[32][33];
  const int bx = blockIdx.x * 32;          // output-col block (d_out)
  const int by = blockIdx.y * 32;          // input-row block (d_in)
  const int tx = threadIdx.x & 31;
  const int ty = threadIdx.x >> 5;         // 0..7
  #pragma unroll
  for (int j = 0; j < 32; j += 8)
    tile[ty + j][tx] = W[(size_t)(by + ty + j) * Dc + bx + tx];
  __syncthreads();
  #pragma unroll
  for (int j = 0; j < 32; j += 8)
    Wt[(size_t)(bx + ty + j) * Dc + by + tx] = (_Float16)tile[tx][ty + j];
}

// ---------------------------------------------------------------------------
// LayerNorm over rows of f32 X -> f16 Y. One block (256 thr) per row.
// ---------------------------------------------------------------------------
__global__ __launch_bounds__(256) void ln_rows(
    const float* __restrict__ X, const float* __restrict__ G,
    const float* __restrict__ Bv, _Float16* __restrict__ Y) {
  __shared__ float red[8];
  __shared__ float stat[2];
  const int r = blockIdx.x;
  const int t = threadIdx.x;
  const float4 xv = ((const float4*)(X + (size_t)r * Dc))[t];
  float s = xv.x + xv.y + xv.z + xv.w;
  #pragma unroll
  for (int o = 16; o >= 1; o >>= 1) s += __shfl_xor(s, o, 32);
  if ((t & 31) == 0) red[t >> 5] = s;
  __syncthreads();
  if (t == 0) { float a = 0.f; for (int i = 0; i < 8; ++i) a += red[i];
                stat[0] = a * (1.0f / Dc); }
  __syncthreads();
  const float mu = stat[0];
  const float dx = xv.x - mu, dy = xv.y - mu, dz = xv.z - mu, dw = xv.w - mu;
  float q = dx * dx + dy * dy + dz * dz + dw * dw;
  #pragma unroll
  for (int o = 16; o >= 1; o >>= 1) q += __shfl_xor(q, o, 32);
  if ((t & 31) == 0) red[t >> 5] = q;
  __syncthreads();
  if (t == 0) { float a = 0.f; for (int i = 0; i < 8; ++i) a += red[i];
                stat[1] = rsqrtf(a * (1.0f / Dc) + 1e-5f); }
  __syncthreads();
  const float rs = stat[1];
  const int c = t * 4;
  _Float16* yr = Y + (size_t)r * Dc + c;
  yr[0] = (_Float16)(dx * rs * G[c]     + Bv[c]);
  yr[1] = (_Float16)(dy * rs * G[c + 1] + Bv[c + 1]);
  yr[2] = (_Float16)(dz * rs * G[c + 2] + Bv[c + 2]);
  yr[3] = (_Float16)(dw * rs * G[c + 3] + Bv[c + 3]);
}

// Fused: H = X + Attn (store f32), then LN(H) -> f16 Y.
__global__ __launch_bounds__(256) void add_ln_rows(
    const float* __restrict__ X, const float* __restrict__ At,
    const float* __restrict__ G, const float* __restrict__ Bv,
    float* __restrict__ Hr, _Float16* __restrict__ Y) {
  __shared__ float red[8];
  __shared__ float stat[2];
  const int r = blockIdx.x;
  const int t = threadIdx.x;
  const float4 xv = ((const float4*)(X  + (size_t)r * Dc))[t];
  const float4 av = ((const float4*)(At + (size_t)r * Dc))[t];
  float4 hv; hv.x = xv.x + av.x; hv.y = xv.y + av.y;
             hv.z = xv.z + av.z; hv.w = xv.w + av.w;
  ((float4*)(Hr + (size_t)r * Dc))[t] = hv;
  float s = hv.x + hv.y + hv.z + hv.w;
  #pragma unroll
  for (int o = 16; o >= 1; o >>= 1) s += __shfl_xor(s, o, 32);
  if ((t & 31) == 0) red[t >> 5] = s;
  __syncthreads();
  if (t == 0) { float a = 0.f; for (int i = 0; i < 8; ++i) a += red[i];
                stat[0] = a * (1.0f / Dc); }
  __syncthreads();
  const float mu = stat[0];
  const float dx = hv.x - mu, dy = hv.y - mu, dz = hv.z - mu, dw = hv.w - mu;
  float q = dx * dx + dy * dy + dz * dz + dw * dw;
  #pragma unroll
  for (int o = 16; o >= 1; o >>= 1) q += __shfl_xor(q, o, 32);
  if ((t & 31) == 0) red[t >> 5] = q;
  __syncthreads();
  if (t == 0) { float a = 0.f; for (int i = 0; i < 8; ++i) a += red[i];
                stat[1] = rsqrtf(a * (1.0f / Dc) + 1e-5f); }
  __syncthreads();
  const float rs = stat[1];
  const int c = t * 4;
  _Float16* yr = Y + (size_t)r * Dc + c;
  yr[0] = (_Float16)(dx * rs * G[c]     + Bv[c]);
  yr[1] = (_Float16)(dy * rs * G[c + 1] + Bv[c + 1]);
  yr[2] = (_Float16)(dz * rs * G[c + 2] + Bv[c + 2]);
  yr[3] = (_Float16)(dw * rs * G[c + 3] + Bv[c + 3]);
}

// ---------------------------------------------------------------------------
// WMMA GEMM: C[4096,1024] = A[4096,1024] * Bt^T. A f16 row-major; Bt f16
// (N,K) row-major (pre-transposed weight). Block = 8 waves = 128(M) x 64(N).
// B tile (64 cols x 64 K = 8KB) is staged in LDS once per block per stage,
// double-buffered via the Tensor Data Mover (wave 0 issues tensor_load_to_lds,
// fenced by s_wait_tensorcnt + workgroup barrier); all waves read B-fragments
// from LDS (ds_load_b128 pairs). A-fragments stream straight from global.
// EPI: 0 = plain f16 store, 1 = V head-transpose f16 store [b,h,d,n],
//      2 = +bias, exact-erf GELU, f16 store, 3 = +bias +residual, f32 store.
// ---------------------------------------------------------------------------
template <int EPI>
__global__ __launch_bounds__(256) void gemm_wmma(
    const _Float16* __restrict__ A, const _Float16* __restrict__ Bt,
    void* __restrict__ dstv, const float* __restrict__ bias,
    const float* __restrict__ res) {
  __shared__ __align__(32) _Float16 bs[2][64 * 64];  // 2 x 8KB B stages
  const int lane = threadIdx.x & 31;
  const int w    = threadIdx.x >> 5;
  const int ln   = lane & 15;
  const int lg   = lane >> 4;
  const int m0   = blockIdx.x * 128 + w * 16;
  const int n0   = blockIdx.y * 64;

  const _Float16* arow = A + (size_t)(m0 + ln) * Dc + lg * 16;  // A-frag base
  constexpr int NSTG = Dc / 64;  // 16 stages of K=64

#if USE_TDM
  if (w == 0)
    tdm_load_b_tile(Bt + (size_t)n0 * Dc, (unsigned)(uintptr_t)&bs[0][0]);
#endif

  v8f acc[4] = {};
  for (int s = 0; s < NSTG; ++s) {
    const int nb  = s & 1;
    const int kks = s * 64;
#if USE_TDM
    if (w == 0) {
      if (s + 1 < NSTG) {
        tdm_load_b_tile(Bt + (size_t)n0 * Dc + (kks + 64),
                        (unsigned)(uintptr_t)&bs[nb ^ 1][0]);
        __builtin_amdgcn_s_wait_tensorcnt((short)1);  // current stage done
      } else {
        __builtin_amdgcn_s_wait_tensorcnt((short)0);  // last stage done
      }
    }
#else
    {  // cooperative staging fallback: 256 thr x 32B = 8KB
      const int col = threadIdx.x >> 2, chunk = threadIdx.x & 3;
      *(v16h*)&bs[nb][col * 64 + chunk * 16] =
          *(const v16h*)(Bt + (size_t)(n0 + col) * Dc + kks + chunk * 16);
    }
#endif
    __syncthreads();  // B stage visible to all waves
    #pragma unroll
    for (int ks = 0; ks < 2; ++ks) {
      const int kk = kks + ks * 32;
      v16h af = ldA(arow + kk);
      __builtin_prefetch(arow + kk + 256, 0, 0);
      #pragma unroll
      for (int t2 = 0; t2 < 4; ++t2) {
        v16h bf = *(const v16h*)&bs[nb][(t2 * 16 + ln) * 64 + ks * 32 + lg * 16];
        acc[t2] = wmma_f16(af, bf, acc[t2]);
      }
    }
    __syncthreads();  // all reads done before buffer is overwritten
  }

  #pragma unroll
  for (int t2 = 0; t2 < 4; ++t2) {
    const int c = n0 + t2 * 16 + ln;
    #pragma unroll
    for (int v = 0; v < 8; ++v) {
      const int r = m0 + v + lg * 8;
      const float val = acc[t2][v];
      if constexpr (EPI == 0) {
        ((_Float16*)dstv)[(size_t)r * Dc + c] = (_Float16)val;
      } else if constexpr (EPI == 1) {
        const int b = r / Nseq, n = r % Nseq;
        const int h = c / HDc,  d = c % HDc;
        ((_Float16*)dstv)[((size_t)(b * Hc + h) * HDc + d) * Nseq + n] =
            (_Float16)val;
      } else if constexpr (EPI == 2) {
        const float xg = val + bias[c];
        const float g  = 0.5f * xg * (1.0f + erff(xg * 0.70710678118654752f));
        ((_Float16*)dstv)[(size_t)r * Dc + c] = (_Float16)g;
      } else {
        ((float*)dstv)[(size_t)r * Dc + c] =
            val + bias[c] + res[(size_t)r * Dc + c];
      }
    }
  }
}

// ---------------------------------------------------------------------------
// Causal attention (no 1/sqrt(hd) scaling, per reference), flash-style.
// One wave per (b, h, 16-query tile). Computes S^T = K*Q^T so softmax stats
// are per-lane, then O^T = V^T * P^T. Output written with the reference's
// faithful head-reshape bug: row' = h*128 + q/16, col' = (q%16)*64 + d.
// ---------------------------------------------------------------------------
__global__ __launch_bounds__(256) void attn_wmma(
    const _Float16* __restrict__ Q, const _Float16* __restrict__ Kh,
    const _Float16* __restrict__ Vt, float* __restrict__ Ob) {
  const int lane = threadIdx.x & 31;
  const int ln = lane & 15, lg = lane >> 4, kb = lg * 8;
  const int wv = blockIdx.x * 8 + (threadIdx.x >> 5);
  const int qt = wv & (Nseq / 16 - 1);          // 0..127
  const int h  = (wv >> 7) & (Hc - 1);
  const int b  = wv >> 11;
  const int q0 = qt * 16;

  // Q^T B-fragments (lane = query column, K = head-dim): contiguous 32B loads
  const _Float16* qrow = Q + (size_t)(b * Nseq + q0 + ln) * Dc + h * HDc + lg * 16;
  const v16h qf0 = *(const v16h*)(qrow);        // d in [0,32)
  const v16h qf1 = *(const v16h*)(qrow + 32);   // d in [32,64)

  const _Float16* kb0 = Kh + (size_t)b * Nseq * Dc + h * HDc;
  const _Float16* vb0 = Vt + (size_t)(b * Hc + h) * HDc * Nseq;

  v8f o0 = {}, o1 = {}, o2 = {}, o3 = {};
  float mrow = -INFINITY, lrow = 0.0f;

  for (int kt = 0; kt < q0 + 16; kt += 32) {
    // S^T tiles: s0 = keys kt..kt+15, s1 = keys kt+16..kt+31 (cols = queries)
    const _Float16* kr0 = kb0 + (size_t)(kt + ln) * Dc + lg * 16;
    const _Float16* kr1 = kb0 + (size_t)(kt + 16 + ln) * Dc + lg * 16;
    v8f s0 = {}, s1 = {};
    s0 = wmma_f16(ldA(kr0),      qf0, s0);
    s0 = wmma_f16(ldA(kr0 + 32), qf1, s0);
    s1 = wmma_f16(ldA(kr1),      qf0, s1);
    s1 = wmma_f16(ldA(kr1 + 32), qf1, s1);

    if (kt + 31 > q0) {                         // diagonal block: causal mask
      const int qg = q0 + ln;
      #pragma unroll
      for (int v = 0; v < 8; ++v) {
        const int k0g = kt + kb + v;
        if (k0g      > qg) s0[v] = -INFINITY;
        if (k0g + 16 > qg) s1[v] = -INFINITY;
      }
    }

    // Online softmax per query column (lane + its xor-16 partner)
    float bm = -INFINITY;
    #pragma unroll
    for (int v = 0; v < 8; ++v) bm = fmaxf(bm, fmaxf(s0[v], s1[v]));
    bm = fmaxf(bm, __shfl_xor(bm, 16, 32));
    const float mnew  = fmaxf(mrow, bm);
    const float scale = __expf(mrow - mnew);
    float p0[8], p1[8], ps = 0.f;
    #pragma unroll
    for (int v = 0; v < 8; ++v) {
      p0[v] = __expf(s0[v] - mnew);
      p1[v] = __expf(s1[v] - mnew);
      ps += p0[v] + p1[v];
    }
    ps += __shfl_xor(ps, 16, 32);
    lrow = lrow * scale + ps;
    mrow = mnew;
    #pragma unroll
    for (int v = 0; v < 8; ++v) {
      o0[v] *= scale; o1[v] *= scale; o2[v] *= scale; o3[v] *= scale;
    }

    // Repack P^T into a B-fragment (lane needs keys 16*lg + 0..15)
    v16h pf;
    #pragma unroll
    for (int e = 0; e < 8; ++e) {
      const float x0 = __shfl_xor(p0[e], 16, 32);  // partner's s0 rows
      const float x1 = __shfl_xor(p1[e], 16, 32);  // partner's s1 rows
      pf[e]     = (_Float16)(lg ? x1    : p0[e]);
      pf[e + 8] = (_Float16)(lg ? p1[e] : x0);
    }

    // O^T += V^T * P^T  (A = V^T rows = head-dim, contiguous over keys)
    const _Float16* vr = vb0 + (size_t)ln * Nseq + kt + lg * 16;
    o0 = wmma_f16(ldA(vr),                      pf, o0);
    o1 = wmma_f16(ldA(vr + (size_t)16 * Nseq), pf, o1);
    o2 = wmma_f16(ldA(vr + (size_t)32 * Nseq), pf, o2);
    o3 = wmma_f16(ldA(vr + (size_t)48 * Nseq), pf, o3);
  }

  const float ri = 1.0f / lrow;
  // Reference reshape bug: row' = h*128 + qt, col' = (q%16)*64 + d
  float* orow = Ob + ((size_t)b * Nseq + h * (Nseq / 16) + qt) * Dc
                   + (size_t)ln * HDc + kb;
  #pragma unroll
  for (int v = 0; v < 8; ++v) {
    orow[v]      = o0[v] * ri;
    orow[16 + v] = o1[v] * ri;
    orow[32 + v] = o2[v] * ri;
    orow[48 + v] = o3[v] * ri;
  }
}

// ---------------------------------------------------------------------------
extern "C" void kernel_launch(void* const* d_in, const int* in_sizes, int n_in,
                              void* d_out, int out_size, void* d_ws,
                              size_t ws_size, hipStream_t stream) {
  (void)in_sizes; (void)n_in; (void)out_size; (void)ws_size;
  const float* x   = (const float*)d_in[0];
  const float* wq  = (const float*)d_in[1];
  const float* wk  = (const float*)d_in[2];
  const float* wv  = (const float*)d_in[3];
  const float* w1  = (const float*)d_in[4];
  const float* b1  = (const float*)d_in[5];
  const float* w2  = (const float*)d_in[6];
  const float* b2  = (const float*)d_in[7];
  const float* g1  = (const float*)d_in[8];
  const float* be1 = (const float*)d_in[9];
  const float* g2  = (const float*)d_in[10];
  const float* be2 = (const float*)d_in[11];
  float* out = (float*)d_out;

  char* ws = (char*)d_ws;
  size_t off = 0;
  auto take = [&](size_t bytes) -> char* {
    char* p = ws + off;
    off += (bytes + 255) & ~(size_t)255;
    return p;
  };
  const size_t WH = (size_t)Dc * Dc * sizeof(_Float16);   // 2 MB
  const size_t AH = (size_t)BNc * Dc * sizeof(_Float16);  // 8 MB
  const size_t AF = (size_t)BNc * Dc * sizeof(float);     // 16 MB
  _Float16* wqT = (_Float16*)take(WH);
  _Float16* wkT = (_Float16*)take(WH);
  _Float16* wvT = (_Float16*)take(WH);
  _Float16* w1T = (_Float16*)take(WH);
  _Float16* w2T = (_Float16*)take(WH);
  _Float16* xn  = (_Float16*)take(AH);   // LN1(x) f16
  _Float16* qh  = (_Float16*)take(AH);   // Q f16 [bn, d]
  _Float16* kh  = (_Float16*)take(AH);   // K f16 [bn, d]
  _Float16* vt  = (_Float16*)take(AH);   // V f16 [b, h, hd, n]
  float*    att = (float*)take(AF);      // attention out (bug-scattered)
  float*    hres= (float*)take(AF);      // h = x + attn
  _Float16* y2  = (_Float16*)take(AH);   // LN2(h) f16
  _Float16* ffh = (_Float16*)take(AH);   // GELU(h@w1+b1) f16

  const dim3 blk(256);
  const dim3 tgrid(Dc / 32, Dc / 32);
  w_transpose_f16<<<tgrid, blk, 0, stream>>>(wq, wqT);
  w_transpose_f16<<<tgrid, blk, 0, stream>>>(wk, wkT);
  w_transpose_f16<<<tgrid, blk, 0, stream>>>(wv, wvT);
  w_transpose_f16<<<tgrid, blk, 0, stream>>>(w1, w1T);
  w_transpose_f16<<<tgrid, blk, 0, stream>>>(w2, w2T);

  ln_rows<<<BNc, blk, 0, stream>>>(x, g1, be1, xn);

  const dim3 ggrid(BNc / 128, Dc / 64);
  gemm_wmma<0><<<ggrid, blk, 0, stream>>>(xn, wqT, qh, nullptr, nullptr);
  gemm_wmma<0><<<ggrid, blk, 0, stream>>>(xn, wkT, kh, nullptr, nullptr);
  gemm_wmma<1><<<ggrid, blk, 0, stream>>>(xn, wvT, vt, nullptr, nullptr);

  const int attn_blocks = (2 * Hc * (Nseq / 16)) / 8;  // 512
  attn_wmma<<<attn_blocks, blk, 0, stream>>>(qh, kh, vt, att);

  add_ln_rows<<<BNc, blk, 0, stream>>>(x, att, g2, be2, hres, y2);

  gemm_wmma<2><<<ggrid, blk, 0, stream>>>(y2, w1T, ffh, b1, nullptr);
  gemm_wmma<3><<<ggrid, blk, 0, stream>>>(ffh, w2T, out, b2, hres);
}
